// EncoderText_68040871903736
// MI455X (gfx1250) — compile-verified, matrix operand order checked by reading, not run on hardware
//
#include <hip/hip_runtime.h>

// ---------------------------------------------------------------------------
// SRU text encoder for MI455X (gfx1250, wave32, WMMA + async global->LDS).
//   K=620 (pad 640), D=2400, B=128, T=60, R=T*B=7680 rows.
// GEMMs (0.89 TFLOP) run on v_wmma_f32_16x16x32_bf16 with double-buffered
// global_load_async_to_lds_b128 staging (ASYNCcnt pipelined). W is
// pre-transposed so A and B tiles stage/fragment identically.
// ---------------------------------------------------------------------------

#define KDIM   620
#define KPAD   640
#define DDIM   2400
#define BDIM   128
#define TDIM   60
#define RROWS  (TDIM * BDIM)   // 7680

typedef __attribute__((ext_vector_type(16))) __bf16 v16bf;
typedef __attribute__((ext_vector_type(8)))  float  v8f;

union ABFrag { v16bf v; unsigned int u[8]; };

__device__ __forceinline__ unsigned short f2bf(float f) {
    unsigned int u = __float_as_uint(f);
    u += 0x7FFFu + ((u >> 16) & 1u);      // round-to-nearest-even
    return (unsigned short)(u >> 16);
}

// Async copy 16B global -> LDS (per-lane addresses), tracked by ASYNCcnt.
// Low 32 bits of a flat shared-memory pointer are the LDS byte offset.
__device__ __forceinline__ void async_copy_b128(void* lds_ptr, const void* gptr) {
    asm volatile("global_load_async_to_lds_b128 %0, %1, off"
                 :: "v"((unsigned int)(unsigned long long)lds_ptr),
                    "v"((unsigned long long)gptr)
                 : "memory");
}

// ---------------------------------------------------------------------------
// 1) Embedding gather: tokens -> time-major bf16 activation matrix [R x KPAD]
// ---------------------------------------------------------------------------
__global__ void embed_gather_kernel(const int* __restrict__ x,
                                    const float* __restrict__ embed,
                                    unsigned short* __restrict__ Xbf) {
    const int r = blockIdx.x;                 // r = t*B + b (time-major)
    const int t = r / BDIM, b = r % BDIM;
    const int tok = x[b * TDIM + t];
    const float* e = embed + (size_t)tok * KDIM;
    for (int c = threadIdx.x; c < KPAD; c += blockDim.x) {
        float v = (c < KDIM) ? e[c] : 0.0f;
        Xbf[(size_t)r * KPAD + c] = f2bf(v);
    }
}

// ---------------------------------------------------------------------------
// 2) Weight transpose + f32->bf16:  Wt[n*Kpad + k] = bf16(W[k*Out + n])
//    LDS-tiled 32x32 so both global reads and writes stay coalesced.
// ---------------------------------------------------------------------------
__global__ __launch_bounds__(256) void convert_wt_kernel(
        const float* __restrict__ W, unsigned short* __restrict__ Wt,
        int Kin, int Kpad, int Out) {
    __shared__ float tile[32][33];
    const int k0 = blockIdx.y * 32, n0 = blockIdx.x * 32;
#pragma unroll
    for (int i = threadIdx.y; i < 32; i += 8) {
        int k = k0 + i, n = n0 + threadIdx.x;
        tile[i][threadIdx.x] = (k < Kin && n < Out) ? W[(size_t)k * Out + n] : 0.0f;
    }
    __syncthreads();
#pragma unroll
    for (int i = threadIdx.y; i < 32; i += 8) {
        int n = n0 + i, k = k0 + threadIdx.x;
        if (n < Out && k < Kpad)
            Wt[(size_t)n * Kpad + k] = f2bf(tile[threadIdx.x][i]);
    }
}

// ---------------------------------------------------------------------------
// 3) Tiled bf16 WMMA GEMM:  C[M x Out] f32 = X[M x Kp] * Wt[Out x Kp]^T
//    Block = 256 threads = 8 waves (4m x 2n); block tile 128M x 128N;
//    wave tile 32M x 64N -> 8 WMMAs per 32-wide K step.
//    Double-buffered async global->LDS staging: prefetch tile k+1 while
//    computing tile k; s_wait_asynccnt 0x4 retires the current tile's copies.
// ---------------------------------------------------------------------------
__global__ __launch_bounds__(256) void gemm_bf16_wmma(
        const unsigned short* __restrict__ Xbf, int ldx,
        const unsigned short* __restrict__ Wt, int Kp, int Out,
        float* __restrict__ C) {
    // Two buffers, each: A tile 128x32 bf16 (2048 dw) + B tile 128x32 (2048 dw)
    __shared__ unsigned int smem[2 * 4096];

    const int tid   = threadIdx.x;
    const int lane  = tid & 31;
    const int wave  = tid >> 5;
    const int wm0   = (wave >> 1) * 32;       // wave M offset in block tile
    const int wn0   = (wave & 1)  * 64;       // wave N offset in block tile
    const int m0    = blockIdx.y * 128;
    const int n0    = blockIdx.x * 128;
    const int r15   = lane & 15;
    const int khalf = lane >> 4;              // lanes 16-31 hold K 8-15 and 24-31

    // Staging assignment: each thread copies 2 x 16B of A and 2 x 16B of B.
    const int srow0 = tid >> 2, sq0 = tid & 3;              // chunks 0..255
    const int srow1 = (tid + 256) >> 2, sq1 = (tid + 256) & 3; // chunks 256..511
    int nrow0 = n0 + srow0; if (nrow0 >= Out) nrow0 = Out - 1; // clamp, never skip
    int nrow1 = n0 + srow1; if (nrow1 >= Out) nrow1 = Out - 1;

    v8f acc[2][4] = {};
    const int nsteps = Kp >> 5;               // Kp % 32 == 0

    // Prologue: stage tile 0 into buffer 0.
    {
        unsigned int* bufA = smem;
        unsigned int* bufB = smem + 2048;
        async_copy_b128(bufA + srow0 * 16 + sq0 * 4,
                        Xbf + (size_t)(m0 + srow0) * ldx + sq0 * 8);
        async_copy_b128(bufB + srow0 * 16 + sq0 * 4,
                        Wt + (size_t)nrow0 * Kp + sq0 * 8);
        async_copy_b128(bufA + srow1 * 16 + sq1 * 4,
                        Xbf + (size_t)(m0 + srow1) * ldx + sq1 * 8);
        async_copy_b128(bufB + srow1 * 16 + sq1 * 4,
                        Wt + (size_t)nrow1 * Kp + sq1 * 8);
    }

    for (int step = 0; step < nsteps; ++step) {
        const int s = step & 1;
        if (step + 1 < nsteps) {
            // Prefetch next K tile into the other buffer (4 async ops/wave).
            const int kk = (step + 1) << 5;
            unsigned int* bufA = smem + (s ^ 1) * 4096;
            unsigned int* bufB = bufA + 2048;
            async_copy_b128(bufA + srow0 * 16 + sq0 * 4,
                            Xbf + (size_t)(m0 + srow0) * ldx + kk + sq0 * 8);
            async_copy_b128(bufB + srow0 * 16 + sq0 * 4,
                            Wt + (size_t)nrow0 * Kp + kk + sq0 * 8);
            async_copy_b128(bufA + srow1 * 16 + sq1 * 4,
                            Xbf + (size_t)(m0 + srow1) * ldx + kk + sq1 * 8);
            async_copy_b128(bufB + srow1 * 16 + sq1 * 4,
                            Wt + (size_t)nrow1 * Kp + kk + sq1 * 8);
            // 8 outstanding; waiting to <=4 retires the current tile (in-order).
            asm volatile("s_wait_asynccnt 0x4" ::: "memory");
        } else {
            asm volatile("s_wait_asynccnt 0x0" ::: "memory");
        }
        __syncthreads();

        const unsigned int* bufA = smem + s * 4096;
        const unsigned int* bufB = bufA + 2048;

        // Fragments per ISA 16-bit 16x32 layout: per lane, two contiguous
        // dword quads at row*16 + khalf*4 and row*16 + 8 + khalf*4.
        ABFrag a[2], bfr[4];
#pragma unroll
        for (int mt = 0; mt < 2; ++mt) {
            const int row = wm0 + mt * 16 + r15;
#pragma unroll
            for (int j = 0; j < 8; ++j) {
                const int off = (j < 4 ? j : 8 + (j - 4)) + khalf * 4;
                a[mt].u[j] = bufA[row * 16 + off];
            }
        }
#pragma unroll
        for (int nt = 0; nt < 4; ++nt) {
            const int col = wn0 + nt * 16 + r15;
#pragma unroll
            for (int j = 0; j < 8; ++j) {
                const int off = (j < 4 ? j : 8 + (j - 4)) + khalf * 4;
                bfr[nt].u[j] = bufB[col * 16 + off];
            }
        }

#pragma unroll
        for (int mt = 0; mt < 2; ++mt)
#pragma unroll
            for (int nt = 0; nt < 4; ++nt)
                acc[mt][nt] = __builtin_amdgcn_wmma_f32_16x16x32_bf16(
                    false, a[mt].v, false, bfr[nt].v,
                    (short)0, acc[mt][nt], false, false);
        __syncthreads();   // all reads of buffer s done before it is restaged
    }

    // Store per 32-bit C/D layout: VGPR g -> M = khalf*8 + g, N = lane&15.
#pragma unroll
    for (int mt = 0; mt < 2; ++mt)
#pragma unroll
        for (int nt = 0; nt < 4; ++nt) {
            const int col = n0 + wn0 + nt * 16 + r15;
            if (col < Out) {
#pragma unroll
                for (int g = 0; g < 8; ++g) {
                    const int row = m0 + wm0 + mt * 16 + khalf * 8 + g;
                    C[(size_t)row * Out + col] = acc[mt][nt][g];
                }
            }
        }
}

// ---------------------------------------------------------------------------
// 4) SRU gated scan (sequential over T, parallel over B*D, coalesced in d).
//    Writes f32 h (next layer's highway input) and bf16 h (next GEMM's A).
// ---------------------------------------------------------------------------
__global__ void sru_scan_kernel(const float* __restrict__ U, int kD, int kparts,
                                const float* __restrict__ bias,
                                const float* __restrict__ Hin,
                                float* __restrict__ Hout,
                                unsigned short* __restrict__ Xnext) {
    const int idx = blockIdx.x * blockDim.x + threadIdx.x;
    if (idx >= BDIM * DDIM) return;
    const int d = idx % DDIM, b = idx / DDIM;
    const float bf = bias[d], br = bias[DDIM + d];
    float c = 0.0f;
    for (int t = 0; t < TDIM; ++t) {
        const size_t rowU = (size_t)(t * BDIM + b) * kD;
        const float cand = U[rowU + d];
        const float fg = 1.0f / (1.0f + __expf(-(U[rowU + DDIM + d] + bf)));
        const float rg = 1.0f / (1.0f + __expf(-(U[rowU + 2 * DDIM + d] + br)));
        const float xr = (kparts == 4) ? U[rowU + 3 * DDIM + d]
                                       : Hin[(size_t)(t * BDIM + b) * DDIM + d];
        c = fg * c + (1.0f - fg) * cand;
        const float h = rg * tanhf(c) + (1.0f - rg) * xr;
        const size_t ro = (size_t)(t * BDIM + b) * DDIM + d;
        Hout[ro] = h;
        Xnext[ro] = f2bf(h);
    }
}

// ---------------------------------------------------------------------------
// 5) Gather last valid timestep + L2 normalize (one block per batch element)
// ---------------------------------------------------------------------------
__global__ void final_norm_kernel(const float* __restrict__ H,
                                  const int* __restrict__ lengths,
                                  float* __restrict__ out) {
    __shared__ float red[256];
    const int b = blockIdx.x;
    int t = lengths[b] - 1;
    t = t < 0 ? 0 : (t >= TDIM ? TDIM - 1 : t);
    const float* h = H + (size_t)(t * BDIM + b) * DDIM;
    float s = 0.0f;
    for (int d = threadIdx.x; d < DDIM; d += 256) { float v = h[d]; s += v * v; }
    red[threadIdx.x] = s;
    __syncthreads();
    for (int o = 128; o > 0; o >>= 1) {
        if (threadIdx.x < o) red[threadIdx.x] += red[threadIdx.x + o];
        __syncthreads();
    }
    const float inv = rsqrtf(red[0]);
    for (int d = threadIdx.x; d < DDIM; d += 256)
        out[(size_t)b * DDIM + d] = h[d] * inv;
}

// ---------------------------------------------------------------------------
extern "C" void kernel_launch(void* const* d_in, const int* in_sizes, int n_in,
                              void* d_out, int out_size, void* d_ws, size_t ws_size,
                              hipStream_t stream) {
    (void)in_sizes; (void)n_in; (void)out_size; (void)ws_size;
    const int*   x       = (const int*)  d_in[0];
    const int*   lengths = (const int*)  d_in[1];
    const float* embed   = (const float*)d_in[2];
    const float* W0 = (const float*)d_in[3];
    const float* b0 = (const float*)d_in[4];
    const float* W1 = (const float*)d_in[5];
    const float* b1 = (const float*)d_in[6];
    const float* W2 = (const float*)d_in[7];
    const float* b2 = (const float*)d_in[8];
    const float* W3 = (const float*)d_in[9];
    const float* b3 = (const float*)d_in[10];
    float* out = (float*)d_out;

    // Workspace layout (all sizes 256B-aligned):
    char* ws = (char*)d_ws;
    size_t off = 0;
    unsigned short* Xbf = (unsigned short*)(ws + off); off += (size_t)RROWS * DDIM * 2;    // 36.9 MB
    unsigned short* Wt  = (unsigned short*)(ws + off); off += (size_t)3 * DDIM * DDIM * 2; // 34.6 MB (max layer)
    float* U  = (float*)(ws + off); off += (size_t)RROWS * 4 * DDIM * 4;                   // 295 MB
    float* H0 = (float*)(ws + off); off += (size_t)RROWS * DDIM * 4;                       // 73.7 MB
    float* H1 = (float*)(ws + off);                                                        // 73.7 MB

    // ---- Layer 0 (K=620 -> pad 640 in, 4*D out) ----
    embed_gather_kernel<<<RROWS, 256, 0, stream>>>(x, embed, Xbf);
    {
        dim3 tgrid((4 * DDIM + 31) / 32, (KPAD + 31) / 32);
        convert_wt_kernel<<<tgrid, dim3(32, 8), 0, stream>>>(W0, Wt, KDIM, KPAD, 4 * DDIM);
        dim3 grid((4 * DDIM + 127) / 128, RROWS / 128);
        gemm_bf16_wmma<<<grid, 256, 0, stream>>>(Xbf, KPAD, Wt, KPAD, 4 * DDIM, U);
    }
    sru_scan_kernel<<<(BDIM * DDIM + 255) / 256, 256, 0, stream>>>(
        U, 4 * DDIM, 4, b0, nullptr, H0, Xbf);

    // ---- Layers 1..3 (D in, 3*D out), ping-pong H buffers ----
    const float* Ws[3] = {W1, W2, W3};
    const float* bs[3] = {b1, b2, b3};
    float* Hin = H0; float* Hout = H1;
    for (int l = 0; l < 3; ++l) {
        dim3 tgrid((3 * DDIM + 31) / 32, (DDIM + 31) / 32);
        convert_wt_kernel<<<tgrid, dim3(32, 8), 0, stream>>>(Ws[l], Wt, DDIM, DDIM, 3 * DDIM);
        dim3 grid((3 * DDIM + 127) / 128, RROWS / 128);
        gemm_bf16_wmma<<<grid, 256, 0, stream>>>(Xbf, DDIM, Wt, DDIM, 3 * DDIM, U);
        sru_scan_kernel<<<(BDIM * DDIM + 255) / 256, 256, 0, stream>>>(
            U, 3 * DDIM, 3, bs[l], Hin, Hout, Xbf);
        float* tmp = Hin; Hin = Hout; Hout = tmp;
    }

    // Result of last layer lives in Hin after the final swap.
    final_norm_kernel<<<BDIM, 256, 0, stream>>>(Hin, lengths, out);
}
